// Net_13589276525191
// MI455X (gfx1250) — matrix operand order repaired and checked by reading, not compiled, not run on hardware
//
#include <hip/hip_runtime.h>
#include <math.h>

typedef __attribute__((ext_vector_type(16))) _Float16 v16h;
typedef __attribute__((ext_vector_type(8)))  float    v8f;

#define B_GRAPHS 128
#define E_TOTAL  (128 * 2048)
#define NF       128   // F_IN == NHID == 128

// ---------------- workspace layout (bytes) ----------------
#define OFF_H      0ull                       // 65536*128*4 = 32 MiB
#define OFF_AGG    33554432ull                // 32 MiB
#define OFF_XP     67108864ull                // pooled X, 16 MiB max
#define OFF_SCORE  83886080ull                // 65536 f32
#define OFF_NEWID  84148224ull                // 65536 i32
#define OFF_DEG    84410368ull                // 65536 f32
#define OFF_DINV   84672512ull                // 65536 f32
#define OFF_SRCA   84934656ull                // E i32
#define OFF_DSTA   85983232ull
#define OFF_EWA    87031808ull
#define OFF_SRCB   88080384ull
#define OFF_DSTB   89128960ull
#define OFF_EWB    90177536ull
#define OFF_WF     91226112ull                // 3 * 128*128 f16 fragments
#define OFF_Z      91357184ull                // 128*256 f32
#define OFF_T1     91488256ull                // 128*128 f32
#define OFF_T2     91553792ull                // 128*64 f32
#define OFF_PN     91586560ull                // 1 f32 (1/||p||)

// ---------------- tiny utility kernels ----------------
__global__ void fill_f32(float* __restrict__ p, float v, int n) {
    int i = blockIdx.x * blockDim.x + threadIdx.x;
    if (i < n) p[i] = v;
}

// Pack W (128x128 row-major, W[k*128+n]) into WMMA B-fragment order:
// Wf[((nt*4+ks)*32 + lane)*16 + i] = (f16) W[(ks*32 + (lane>>4)*8 + (i&7) + (i>>3)*16)*128 + nt*16 + (lane&15)]
__global__ void pack_w_frag(const float* __restrict__ W, _Float16* __restrict__ Wf) {
    int idx = blockIdx.x * blockDim.x + threadIdx.x;   // 8*4*32*16 = 16384
    if (idx >= 16384) return;
    int i    = idx & 15;
    int lane = (idx >> 4) & 31;
    int ks   = (idx >> 9) & 3;
    int nt   = idx >> 11;
    int n = nt * 16 + (lane & 15);
    int k = ks * 32 + (lane >> 4) * 8 + (i & 7) + (i >> 3) * 16;
    Wf[idx] = (_Float16)W[k * NF + n];
}

// ---------------- WMMA GEMM: H(Mx128) = X(Mx128) @ W(128x128) ----------------
// block = 256 threads = 8 waves; each wave -> one 16-row strip; grid = M/128.
__global__ void gemm128_wmma(const float* __restrict__ X, const _Float16* __restrict__ Wf,
                             float* __restrict__ H, int M) {
    int wave = threadIdx.x >> 5;
    int lane = threadIdx.x & 31;
    int mBase = blockIdx.x * 128 + wave * 16;
    if (mBase >= M) return;                 // uniform per-wave; M is a multiple of 128
    int r = lane & 15;
    int half = lane >> 4;
    const float* xrow = X + (size_t)(mBase + r) * NF;

    // A fragments for 4 K-steps (16-bit A 16x32 layout, ISA 7.12.2)
    v16h afrag[4];
#pragma unroll
    for (int ks = 0; ks < 4; ++ks) {
        int k0 = ks * 32 + half * 8;
#pragma unroll
        for (int i = 0; i < 8; ++i) {
            afrag[ks][i]     = (_Float16)xrow[k0 + i];
            afrag[ks][i + 8] = (_Float16)xrow[k0 + 16 + i];
        }
    }
#pragma unroll
    for (int nt = 0; nt < 8; ++nt) {
        v8f c = {};
#pragma unroll
        for (int ks = 0; ks < 4; ++ks) {
            v16h b = *(const v16h*)(Wf + ((size_t)(nt * 4 + ks) * 32 + lane) * 16);
            c = __builtin_amdgcn_wmma_f32_16x16x32_f16(
                    false, afrag[ks], false, b, (short)0, c, false, false);
        }
        int n = nt * 16 + r;
#pragma unroll
        for (int j = 0; j < 8; ++j)          // C/D layout: VGPR j -> row j + half*8
            H[(size_t)(mBase + half * 8 + j) * NF + n] = c[j];
    }
}

// ---------------- GCN normalization / aggregation ----------------
__global__ void deg_accum(const int* __restrict__ dst, const float* __restrict__ ew,
                          float* __restrict__ deg, int E) {
    int e = blockIdx.x * blockDim.x + threadIdx.x;
    if (e >= E) return;
    float w = ew[e];
    if (w != 0.f) atomicAdd(deg + dst[e], w);
}

__global__ void rsqrt_k(const float* __restrict__ deg, float* __restrict__ dinv, int n) {
    int i = blockIdx.x * blockDim.x + threadIdx.x;
    if (i < n) dinv[i] = rsqrtf(deg[i]);
}

__global__ void agg_init(const float* __restrict__ H, const float* __restrict__ dinv,
                         const float* __restrict__ bias, float* __restrict__ AGG, int nNodes) {
    int idx = blockIdx.x * blockDim.x + threadIdx.x;
    if (idx >= nNodes * NF) return;
    int v = idx >> 7, f = idx & 127;
    float di = dinv[v];
    AGG[idx] = H[idx] * di * di + bias[f];
}

// thread handles (edge, 4-feature chunk): 32 threads per edge
__global__ void msg_scatter(const float* __restrict__ H, const int* __restrict__ src,
                            const int* __restrict__ dst, const float* __restrict__ ew,
                            const float* __restrict__ dinv, float* __restrict__ AGG, int E) {
    int idx = blockIdx.x * blockDim.x + threadIdx.x;
    int e = idx >> 5;
    if (e >= E) return;
    float w = ew[e];
    if (w == 0.f) return;
    int s = src[e], d = dst[e];
    float coeff = dinv[s] * dinv[d] * w;
    int q = (idx & 31) * 4;
    const float4 h4 = *(const float4*)(H + (size_t)s * NF + q);
    float* out = AGG + (size_t)d * NF + q;
    atomicAdd(out + 0, h4.x * coeff);
    atomicAdd(out + 1, h4.y * coeff);
    atomicAdd(out + 2, h4.z * coeff);
    atomicAdd(out + 3, h4.w * coeff);
}

__global__ void relu_k(float* __restrict__ p, int n) {
    int i = blockIdx.x * blockDim.x + threadIdx.x;
    if (i < n) p[i] = fmaxf(p[i], 0.f);
}

// ---------------- top-k pooling ----------------
__global__ void pnorm_k(const float* __restrict__ p, float* __restrict__ out) {
    float s = 0.f;
    for (int f = 0; f < NF; ++f) s += p[f] * p[f];
    *out = rsqrtf(s);
}

__global__ void score_k(const float* __restrict__ H, const float* __restrict__ p,
                        const float* __restrict__ inv_norm, float* __restrict__ score, int nNodes) {
    int v = blockIdx.x * blockDim.x + threadIdx.x;
    if (v >= nNodes) return;
    const float* h = H + (size_t)v * NF;
    float acc = 0.f;
    for (int f = 0; f < NF; ++f) acc += h[f] * p[f];
    score[v] = acc * (*inv_norm);
}

// One block per graph, n threads. Rank = exact descending sort position
// (tie-break: lower index first) == jax.lax.top_k relabel order.
__global__ void topk_pool(const float* __restrict__ Hrelu, const float* __restrict__ score,
                          float* __restrict__ Xout, int* __restrict__ new_id, int n, int k) {
    __shared__ float s[512];
    int g = blockIdx.x;
    int i = threadIdx.x;
    float si = score[g * n + i];
    s[i] = si;
    __syncthreads();
    int rank = 0;
    for (int j = 0; j < n; ++j) {
        float sj = s[j];
        rank += (sj > si) || (sj == si && j < i);
    }
    if (rank < k) {
        new_id[g * n + i] = g * k + rank;
        float t = tanhf(si);
        const float* hin = Hrelu + (size_t)(g * n + i) * NF;
        float* xo = Xout + (size_t)(g * k + rank) * NF;
        for (int f = 0; f < NF; ++f) xo[f] = hin[f] * t;
    } else {
        new_id[g * n + i] = -1;
    }
}

__global__ void relabel_edges(const int* __restrict__ src, const int* __restrict__ dst,
                              const float* __restrict__ ew, const int* __restrict__ new_id,
                              int* __restrict__ nsrc, int* __restrict__ ndst,
                              float* __restrict__ new_ew, int E) {
    int e = blockIdx.x * blockDim.x + threadIdx.x;
    if (e >= E) return;
    int ns = new_id[src[e]];
    int nd = new_id[dst[e]];
    bool keep = (ns >= 0) && (nd >= 0);
    nsrc[e] = ns > 0 ? ns : 0;
    ndst[e] = nd > 0 ? nd : 0;
    new_ew[e] = keep ? ew[e] : 0.f;
}

// ---------------- readout + head ----------------
__global__ void readout_add(const float* __restrict__ Xp, float* __restrict__ z, int k) {
    int idx = blockIdx.x * blockDim.x + threadIdx.x;   // 128 graphs * 128 feats
    if (idx >= B_GRAPHS * NF) return;
    int g = idx >> 7, f = idx & 127;
    const float* base = Xp + (size_t)(g * k) * NF + f;
    float mx = -3.402823466e38f, sum = 0.f;
    for (int j = 0; j < k; ++j) {
        float v = base[(size_t)j * NF];
        mx = fmaxf(mx, v);
        sum += v;
    }
    z[g * 256 + f]       += mx;
    z[g * 256 + 128 + f] += sum / (float)k;
}

__global__ void linear_relu_k(const float* __restrict__ A, const float* __restrict__ W,
                              const float* __restrict__ bias, float* __restrict__ out,
                              int M, int K, int N) {
    int idx = blockIdx.x * blockDim.x + threadIdx.x;
    if (idx >= M * N) return;
    int m = idx / N, n = idx % N;
    float acc = bias[n];
    const float* a = A + (size_t)m * K;
    for (int k = 0; k < K; ++k) acc += a[k] * W[k * N + n];
    out[idx] = fmaxf(acc, 0.f);
}

__global__ void head3_lsm(const float* __restrict__ t2, const float* __restrict__ L3,
                          const float* __restrict__ bl3, float* __restrict__ out) {
    int row = blockIdx.x * blockDim.x + threadIdx.x;
    if (row >= B_GRAPHS) return;
    const float* t = t2 + (size_t)row * 64;
    float z0 = bl3[0], z1 = bl3[1];
    for (int k = 0; k < 64; ++k) {
        z0 += t[k] * L3[k * 2 + 0];
        z1 += t[k] * L3[k * 2 + 1];
    }
    float m = fmaxf(z0, z1);
    float lse = m + logf(expf(z0 - m) + expf(z1 - m));
    out[row * 2 + 0] = z0 - lse;
    out[row * 2 + 1] = z1 - lse;
}

// ---------------- driver ----------------
extern "C" void kernel_launch(void* const* d_in, const int* in_sizes, int n_in,
                              void* d_out, int out_size, void* d_ws, size_t ws_size,
                              hipStream_t stream) {
    (void)in_sizes; (void)n_in; (void)out_size; (void)ws_size;
    const float* x    = (const float*)d_in[0];
    const int*   src0 = (const int*)d_in[1];
    const int*   dst0 = (const int*)d_in[2];
    const float* W[3]  = {(const float*)d_in[3], (const float*)d_in[5], (const float*)d_in[7]};
    const float* bb[3] = {(const float*)d_in[4], (const float*)d_in[6], (const float*)d_in[8]};
    const float* pp[3] = {(const float*)d_in[9], (const float*)d_in[10], (const float*)d_in[11]};
    const float* L1 = (const float*)d_in[12]; const float* bl1 = (const float*)d_in[13];
    const float* L2 = (const float*)d_in[14]; const float* bl2 = (const float*)d_in[15];
    const float* L3 = (const float*)d_in[16]; const float* bl3 = (const float*)d_in[17];

    char* ws = (char*)d_ws;
    float* H     = (float*)(ws + OFF_H);
    float* AGG   = (float*)(ws + OFF_AGG);
    float* XP    = (float*)(ws + OFF_XP);
    float* SCORE = (float*)(ws + OFF_SCORE);
    int*   NEWID = (int*)  (ws + OFF_NEWID);
    float* DEG   = (float*)(ws + OFF_DEG);
    float* DINV  = (float*)(ws + OFF_DINV);
    int*   SRCA  = (int*)  (ws + OFF_SRCA);
    int*   DSTA  = (int*)  (ws + OFF_DSTA);
    float* EWA   = (float*)(ws + OFF_EWA);
    int*   SRCB  = (int*)  (ws + OFF_SRCB);
    int*   DSTB  = (int*)  (ws + OFF_DSTB);
    float* EWB   = (float*)(ws + OFF_EWB);
    _Float16* WF = (_Float16*)(ws + OFF_WF);
    float* Z     = (float*)(ws + OFF_Z);
    float* T1    = (float*)(ws + OFF_T1);
    float* T2    = (float*)(ws + OFF_T2);
    float* PN    = (float*)(ws + OFF_PN);

    const int E = E_TOTAL;
    const int nper[3] = {512, 256, 128};
    const int kk[3]   = {256, 128, 64};

    // one-time (per call) prep: pack weights to f16 fragments; ew=1; z=0
    for (int s = 0; s < 3; ++s)
        pack_w_frag<<<64, 256, 0, stream>>>(W[s], WF + (size_t)s * 16384);
    fill_f32<<<(E + 255) / 256, 256, 0, stream>>>(EWA, 1.0f, E);
    fill_f32<<<(B_GRAPHS * 256 + 255) / 256, 256, 0, stream>>>(Z, 0.f, B_GRAPHS * 256);

    const float* Xcur = x;
    const int* scur = src0;
    const int* dcur = dst0;
    const float* ecur = EWA;

    for (int s = 0; s < 3; ++s) {
        int M = B_GRAPHS * nper[s];          // nodes this stage (multiple of 128)
        gemm128_wmma<<<M / 128, 256, 0, stream>>>(Xcur, WF + (size_t)s * 16384, H, M);
        fill_f32<<<(M + 255) / 256, 256, 0, stream>>>(DEG, 1.0f, M);
        deg_accum<<<(E + 255) / 256, 256, 0, stream>>>(dcur, ecur, DEG, E);
        rsqrt_k<<<(M + 255) / 256, 256, 0, stream>>>(DEG, DINV, M);
        agg_init<<<(M * NF + 255) / 256, 256, 0, stream>>>(H, DINV, bb[s], AGG, M);
        msg_scatter<<<(E * 32 + 255) / 256, 256, 0, stream>>>(H, scur, dcur, ecur, DINV, AGG, E);
        relu_k<<<(M * NF + 255) / 256, 256, 0, stream>>>(AGG, M * NF);
        pnorm_k<<<1, 1, 0, stream>>>(pp[s], PN);
        score_k<<<(M + 255) / 256, 256, 0, stream>>>(AGG, pp[s], PN, SCORE, M);
        topk_pool<<<B_GRAPHS, nper[s], 0, stream>>>(AGG, SCORE, XP, NEWID, nper[s], kk[s]);
        readout_add<<<64, 256, 0, stream>>>(XP, Z, kk[s]);
        if (s < 2) {
            int*   ns = (s == 0) ? SRCB : SRCA;
            int*   nd = (s == 0) ? DSTB : DSTA;
            float* ne = (s == 0) ? EWB : EWA;
            relabel_edges<<<(E + 255) / 256, 256, 0, stream>>>(scur, dcur, ecur, NEWID, ns, nd, ne, E);
            scur = ns; dcur = nd; ecur = ne;
            Xcur = XP;
        }
    }

    // head MLP + log_softmax
    linear_relu_k<<<(128 * 128 + 255) / 256, 256, 0, stream>>>(Z, L1, bl1, T1, 128, 256, 128);
    linear_relu_k<<<(128 * 64 + 255) / 256, 256, 0, stream>>>(T1, L2, bl2, T2, 128, 128, 64);
    head3_lsm<<<1, 128, 0, stream>>>(T2, L3, bl3, (float*)d_out);
}
